// RSAR_52931176956365
// MI455X (gfx1250) — compile-verified, matrix operand order checked by reading, not run on hardware
//
#include <hip/hip_runtime.h>
#include <hip/hip_bf16.h>
#include <math.h>

// ---------------------------------------------------------------------------
// RSSM world-model loss on MI455X (gfx1250, wave32).
// GEMMs: bf16 WMMA (v_wmma_f32_16x16x32_bf16), f32 accumulate.
// Block = 256 threads = 8 waves. The block's 16xK A-panel is staged into LDS
// once (async global->LDS via gfx1250 GLOBAL_LOAD_ASYNC_TO_LDS_B128), then
// each wave computes NT 16x16 N-tiles per A fetch (NT=4 wide layers, NT=1
// for the 128-wide distribution heads).
// ---------------------------------------------------------------------------

#define TT 64
#define BB 256
#define EMB_ 1024
#define ACT_ 6
#define STOCH 64
#define HID 512
#define FEAT 512
#define STATE 576          // HID + STOCH
#define GK 96              // GRU input K padded: 70 -> 96
#define CATK 1536          // HID + EMB_ (st-net input width)
#define NROW (TT*BB)       // 16384
#define CH 2048            // final-head row chunk
#define APAD 8             // LDS row pad (elements) to break bank conflicts
#define LOG2PI 1.8378770664093453f

typedef __bf16 bf16;
typedef __attribute__((ext_vector_type(16))) __bf16 v16bf;
typedef __attribute__((ext_vector_type(8)))  float  v8f;
typedef int v4i_ __attribute__((vector_size(16)));   // matches builtin param type

#if defined(__has_builtin)
#if __has_builtin(__builtin_amdgcn_global_load_async_to_lds_b128) && \
    __has_builtin(__builtin_amdgcn_s_wait_asynccnt)
#define USE_ASYNC_LDS 1
#endif
#endif
#ifndef USE_ASYNC_LDS
#define USE_ASYNC_LDS 0
#endif

__device__ __forceinline__ float softplusf(float x) {
    return x > 20.f ? x : log1pf(__expf(x));
}
__device__ __forceinline__ float eluf(float x) {
    return x > 0.f ? x : (__expf(x) - 1.f);
}
__device__ __forceinline__ float sigmoidf_(float x) {
    return 1.f / (1.f + __expf(-x));
}

// ---------------------------------------------------------------------------
// WMMA GEMM: out = act(A[M,K] * W[N,K]^T + bias)
// grid = (N/(128*NT), M/16), block = 256. Dynamic LDS = 16*(K+APAD)*2 bytes.
// ---------------------------------------------------------------------------
template <int NT>
__global__ void __launch_bounds__(256)
k_gemm(const bf16* __restrict__ A, const bf16* __restrict__ W,
       const float* __restrict__ bias,
       float* __restrict__ outF, int ldF,
       bf16* __restrict__ outB, int ldB,
       int N, int K, int act)
{
    extern __shared__ char smem[];
    const int tid  = threadIdx.x;
    const int wave = tid >> 5;
    const int lane = tid & 31;
    const int ml   = lane & 15;
    const int grp  = lane >> 4;
    const int ldA  = K + APAD;            // padded LDS row stride (elements)

    // ---- stage the 16 x K A-panel into LDS (once per block) ----
    {
        const int chunksPerRow = K >> 3;          // 16B chunks per row
        const int totalChunks  = chunksPerRow << 4;
        for (int cix = tid; cix < totalChunks; cix += 256) {
            const int r  = cix / chunksPerRow;
            const int cc = cix - r * chunksPerRow;
            const bf16* src = A + ((size_t)blockIdx.y * 16 + r) * K + (cc << 3);
            bf16* dst = (bf16*)smem + (size_t)r * ldA + (cc << 3);
#if USE_ASYNC_LDS
            __builtin_amdgcn_global_load_async_to_lds_b128(
                (__attribute__((address_space(1))) v4i_*)src,
                (__attribute__((address_space(3))) v4i_*)dst, 0, 0);
#else
            *(uint4*)dst = *(const uint4*)src;
#endif
        }
#if USE_ASYNC_LDS
        __builtin_amdgcn_s_wait_asynccnt(0);
#endif
        __syncthreads();
    }

    const bf16* AsRow = (const bf16*)smem + (size_t)ml * ldA;
    const int tileBase = blockIdx.x * (8 * NT) + wave * NT;   // first 16-wide N tile
    const bf16* Wr = W + ((size_t)tileBase * 16 + ml) * K;

    v8f c[NT] = {};
    for (int k0 = 0; k0 < K; k0 += 32) {
        // 16-bit A 16x32 layout: halves 0..7 = K[grp*8..], halves 8..15 = K[16+grp*8..]
        union { v16bf v; uint4 q[2]; } ua;
        ua.q[0] = *(const uint4*)(AsRow + k0 + grp * 8);
        ua.q[1] = *(const uint4*)(AsRow + k0 + 16 + grp * 8);
#pragma unroll
        for (int j = 0; j < NT; ++j) {
            union { v16bf v; uint4 q[2]; } ub;
            const bf16* wp = Wr + (size_t)j * 16 * K;
            ub.q[0] = *(const uint4*)(wp + k0 + grp * 8);
            ub.q[1] = *(const uint4*)(wp + k0 + 16 + grp * 8);
            c[j] = __builtin_amdgcn_wmma_f32_16x16x32_bf16(
                       false, ua.v, false, ub.v, (short)0, c[j], false, false);
        }
    }

#pragma unroll
    for (int j = 0; j < NT; ++j) {
        const int n = (tileBase + j) * 16 + ml;
        const float bv = bias ? bias[n] : 0.f;
#pragma unroll
        for (int q = 0; q < 8; ++q) {
            float x = c[j][q] + bv;
            if (act) x = eluf(x);
            const size_t m = (size_t)blockIdx.y * 16 + q + grp * 8;  // C/D: VGPR q -> M=q+8*grp
            if (outF) outF[m * (size_t)ldF + n] = x;
            if (outB) outB[m * (size_t)ldB + n] = (bf16)x;
        }
    }
}

// ---------------------------------------------------------------------------
// Elementwise / data-movement kernels
// ---------------------------------------------------------------------------
__global__ void k_cvt_bf16(const float* __restrict__ in, bf16* __restrict__ out, int n) {
    int i = blockIdx.x * blockDim.x + threadIdx.x;
    if (i < n) out[i] = (bf16)in[i];
}

__global__ void k_cvt_pad(const float* __restrict__ in, bf16* __restrict__ out,
                          int rows, int kin, int kout) {
    int i = blockIdx.x * blockDim.x + threadIdx.x;
    if (i >= rows * kout) return;
    int r = i / kout, c = i % kout;
    out[i] = (bf16)(c < kin ? in[r * kin + c] : 0.f);
}

__global__ void k_init_state(float* h, bf16* hbf, float* s, bf16* sbf, float* accs) {
    int i = blockIdx.x * blockDim.x + threadIdx.x;
    if (i < BB * HID)   { h[i] = 0.f; hbf[i] = (bf16)0.f; }
    if (i < BB * STOCH) { s[i] = 0.f; sbf[i] = (bf16)0.f; }
    if (i < 4) accs[i] = 0.f;
}

__global__ void k_build_xin(const float* __restrict__ s, const float* __restrict__ a,
                            bf16* __restrict__ x) {
    int i = blockIdx.x * blockDim.x + threadIdx.x;
    if (i >= BB * GK) return;
    int r = i / GK, c = i % GK;
    float v = 0.f;
    if (c < STOCH)             v = s[r * STOCH + c];
    else if (c < STOCH + ACT_) v = a[r * ACT_ + (c - STOCH)];
    x[i] = (bf16)v;
}

__global__ void k_build_hscat(const float* __restrict__ h, const float* __restrict__ s,
                              bf16* __restrict__ o) {
    int i = blockIdx.x * blockDim.x + threadIdx.x;
    if (i >= BB * STATE) return;
    int r = i / STATE, c = i % STATE;
    o[i] = (bf16)(c < HID ? h[r * HID + c] : s[r * STOCH + (c - HID)]);
}

__global__ void k_copy_emb(const float* __restrict__ src, bf16* __restrict__ dst) {
    int i = blockIdx.x * blockDim.x + threadIdx.x;
    if (i >= BB * EMB_) return;
    int r = i / EMB_, c = i % EMB_;
    dst[(size_t)r * CATK + c] = (bf16)src[i];
}

__global__ void k_gru_gate(const float* __restrict__ gi, const float* __restrict__ gh,
                           float* __restrict__ h, bf16* __restrict__ hbf,
                           bf16* __restrict__ catP, bf16* __restrict__ catQ,
                           float* __restrict__ stF, bf16* __restrict__ stB) {
    int i = blockIdx.x * blockDim.x + threadIdx.x;
    if (i >= BB * HID) return;
    int r = i / HID, c = i % HID;
    const float* gir = gi + (size_t)r * 3 * HID;
    const float* ghr = gh + (size_t)r * 3 * HID;
    float rg = sigmoidf_(gir[c] + ghr[c]);
    float z  = sigmoidf_(gir[HID + c] + ghr[HID + c]);
    float nn = tanhf(gir[2 * HID + c] + rg * ghr[2 * HID + c]);
    float hn = (1.f - z) * nn + z * h[i];
    h[i] = hn;
    bf16 hb = (bf16)hn;
    hbf[i] = hb;
    catP[(size_t)r * CATK + c] = hb;
    catQ[(size_t)r * CATK + c] = hb;
    stF[(size_t)r * STATE + c] = hn;
    stB[(size_t)r * STATE + c] = hb;
}

__global__ void k_sample_kl(const float* __restrict__ qraw, const float* __restrict__ praw,
                            const float* __restrict__ epst,
                            float* __restrict__ s, bf16* __restrict__ sbf,
                            float* __restrict__ stF, bf16* __restrict__ stB,
                            float* __restrict__ acc) {
    __shared__ float red[256];
    int i = blockIdx.x * blockDim.x + threadIdx.x;
    float kl = 0.f;
    if (i < BB * STOCH) {
        int r = i / STOCH, j = i % STOCH;
        float qmu = qraw[r * 2 * STOCH + j];
        float qst = softplusf(qraw[r * 2 * STOCH + STOCH + j]) + 1e-4f;
        float pmu = praw[r * 2 * STOCH + j];
        float pst = softplusf(praw[r * 2 * STOCH + STOCH + j]) + 1e-4f;
        float sn = qmu + qst * epst[i];
        s[i] = sn;
        bf16 sb = (bf16)sn;
        sbf[i] = sb;
        stF[(size_t)r * STATE + HID + j] = sn;
        stB[(size_t)r * STATE + HID + j] = sb;
        float d = qmu - pmu;
        kl = __logf(pst / qst) + (qst * qst + d * d) / (2.f * pst * pst) - 0.5f;
    }
    red[threadIdx.x] = kl;
    __syncthreads();
    for (int st = 128; st > 0; st >>= 1) {
        if (threadIdx.x < st) red[threadIdx.x] += red[threadIdx.x + st];
        __syncthreads();
    }
    if (threadIdx.x == 0) atomicAdd(acc, red[0]);
}

__global__ void k_sqloss(const float* __restrict__ mu, const float* __restrict__ tgt,
                         int n, float* __restrict__ acc) {
    __shared__ float red[256];
    int i = blockIdx.x * blockDim.x + threadIdx.x;
    float v = 0.f;
    if (i < n) { float d = tgt[i] - mu[i]; v = 0.5f * d * d; }
    red[threadIdx.x] = v;
    __syncthreads();
    for (int st = 128; st > 0; st >>= 1) {
        if (threadIdx.x < st) red[threadIdx.x] += red[threadIdx.x + st];
        __syncthreads();
    }
    if (threadIdx.x == 0) atomicAdd(acc, red[0]);
}

__global__ void k_rhead(const float* __restrict__ e1, const float* __restrict__ w,
                        const float* __restrict__ b, const float* __restrict__ tgt,
                        int rows, float* __restrict__ acc) {
    __shared__ float red[256];
    int r = blockIdx.x * blockDim.x + threadIdx.x;
    float v = 0.f;
    if (r < rows) {
        const float* er = e1 + (size_t)r * FEAT;
        float sum = b[0];
        for (int k = 0; k < FEAT; ++k) sum += er[k] * w[k];
        float d = tgt[r] - sum;
        v = 0.5f * d * d;
    }
    red[threadIdx.x] = v;
    __syncthreads();
    for (int st = 128; st > 0; st >>= 1) {
        if (threadIdx.x < st) red[threadIdx.x] += red[threadIdx.x + st];
        __syncthreads();
    }
    if (threadIdx.x == 0) atomicAdd(acc, red[0]);
}

__global__ void k_final(const float* __restrict__ acc, float* __restrict__ out) {
    float kl = acc[0] / (float)NROW;                                // KL_SCALE = 1
    float el = acc[1] / (float)NROW + 0.5f * LOG2PI * (float)EMB_;
    float rl = acc[2] / (float)NROW + 0.5f * LOG2PI;
    out[0] = kl + el + rl;
}

// ---------------------------------------------------------------------------
// Host orchestration
// ---------------------------------------------------------------------------
static inline dim3 eblocks(size_t n) { return dim3((unsigned)((n + 255) / 256)); }
static inline size_t ashm(int K) { return (size_t)16 * (K + APAD) * 2; }

extern "C" void kernel_launch(void* const* d_in, const int* in_sizes, int n_in,
                              void* d_out, int out_size, void* d_ws, size_t ws_size,
                              hipStream_t stream)
{
    (void)in_sizes; (void)n_in; (void)out_size; (void)ws_size;
    const float* emb    = (const float*)d_in[0];
    const float* action = (const float*)d_in[1];
    const float* reward = (const float*)d_in[2];
    const float* eps    = (const float*)d_in[3];
    const float* gWih = (const float*)d_in[4],  *gbih = (const float*)d_in[5];
    const float* gWhh = (const float*)d_in[6],  *gbhh = (const float*)d_in[7];
    const float* stW0 = (const float*)d_in[8],  *stb0 = (const float*)d_in[9];
    const float* stW1 = (const float*)d_in[10], *stb1 = (const float*)d_in[11];
    const float* stW2 = (const float*)d_in[12], *stb2 = (const float*)d_in[13];
    const float* epW0 = (const float*)d_in[14], *epb0 = (const float*)d_in[15];
    const float* epW1 = (const float*)d_in[16], *epb1 = (const float*)d_in[17];
    const float* epW2 = (const float*)d_in[18], *epb2 = (const float*)d_in[19];
    const float* rpW0 = (const float*)d_in[20], *rpb0 = (const float*)d_in[21];
    const float* rpW1 = (const float*)d_in[22], *rpb1 = (const float*)d_in[23];
    const float* rpW2 = (const float*)d_in[24], *rpb2 = (const float*)d_in[25];

    size_t off = 0;
    auto alloc = [&](size_t bytes) -> void* {
        void* p = (char*)d_ws + off;
        off = (off + bytes + 255) & ~(size_t)255;
        return p;
    };
    bf16* wihp  = (bf16*)alloc((size_t)3 * HID * GK * 2);
    bf16* whhb  = (bf16*)alloc((size_t)3 * HID * HID * 2);
    bf16* stW0b = (bf16*)alloc((size_t)FEAT * CATK * 2);
    bf16* stW1b = (bf16*)alloc((size_t)FEAT * FEAT * 2);
    bf16* stW2b = (bf16*)alloc((size_t)2 * STOCH * FEAT * 2);
    bf16* epW0b = (bf16*)alloc((size_t)FEAT * STATE * 2);
    bf16* epW1b = (bf16*)alloc((size_t)FEAT * FEAT * 2);
    bf16* epW2b = (bf16*)alloc((size_t)EMB_ * FEAT * 2);
    bf16* rpW0b = (bf16*)alloc((size_t)FEAT * STATE * 2);
    bf16* rpW1b = (bf16*)alloc((size_t)FEAT * FEAT * 2);
    float* h   = (float*)alloc((size_t)BB * HID * 4);
    bf16*  hbf = (bf16*)alloc((size_t)BB * HID * 2);
    float* s   = (float*)alloc((size_t)BB * STOCH * 4);
    bf16*  sbf = (bf16*)alloc((size_t)BB * STOCH * 2);
    bf16*  xinb  = (bf16*)alloc((size_t)BB * GK * 2);
    bf16*  hscat = (bf16*)alloc((size_t)BB * STATE * 2);
    float* gi    = (float*)alloc((size_t)BB * 3 * HID * 4);
    float* gh    = (float*)alloc((size_t)BB * 3 * HID * 4);
    bf16*  t0b   = (bf16*)alloc((size_t)BB * FEAT * 2);
    bf16*  t1b   = (bf16*)alloc((size_t)BB * FEAT * 2);
    bf16*  catP  = (bf16*)alloc((size_t)BB * CATK * 2);   // [next_h | pre_emb]
    bf16*  catQ  = (bf16*)alloc((size_t)BB * CATK * 2);   // [next_h | emb_prev]
    bf16*  u0b   = (bf16*)alloc((size_t)BB * FEAT * 2);
    bf16*  u1b   = (bf16*)alloc((size_t)BB * FEAT * 2);
    bf16*  v0b   = (bf16*)alloc((size_t)BB * FEAT * 2);
    bf16*  v1b   = (bf16*)alloc((size_t)BB * FEAT * 2);
    float* praw  = (float*)alloc((size_t)BB * 2 * STOCH * 4);
    float* qraw  = (float*)alloc((size_t)BB * 2 * STOCH * 4);
    float* statesF = (float*)alloc((size_t)NROW * STATE * 4);
    bf16*  statesB = (bf16*)alloc((size_t)NROW * STATE * 2);
    bf16*  e0b   = (bf16*)alloc((size_t)CH * FEAT * 2);
    bf16*  e1b   = (bf16*)alloc((size_t)CH * FEAT * 2);
    float* embmu = (float*)alloc((size_t)CH * EMB_ * 4);
    bf16*  r0b   = (bf16*)alloc((size_t)CH * FEAT * 2);
    float* r1f   = (float*)alloc((size_t)CH * FEAT * 4);
    float* acc   = (float*)alloc(4 * 4);   // [kl, emb, reward, spare]

    auto gemm4 = [&](const bf16* A, const bf16* Wt, const float* b, float* oF, int ldF,
                     bf16* oB, int ldB, int M, int N, int K, int act) {
        k_gemm<4><<<dim3(N / 512, M / 16), 256, ashm(K), stream>>>(
            A, Wt, b, oF, ldF, oB, ldB, N, K, act);
    };
    auto gemm1 = [&](const bf16* A, const bf16* Wt, const float* b, float* oF, int ldF,
                     bf16* oB, int ldB, int M, int N, int K, int act) {
        k_gemm<1><<<dim3(N / 128, M / 16), 256, ashm(K), stream>>>(
            A, Wt, b, oF, ldF, oB, ldB, N, K, act);
    };

    // --- weight conversion ---
    k_cvt_pad<<<eblocks((size_t)3 * HID * GK), 256, 0, stream>>>(gWih, wihp, 3 * HID, STOCH + ACT_, GK);
    k_cvt_bf16<<<eblocks((size_t)3 * HID * HID), 256, 0, stream>>>(gWhh, whhb, 3 * HID * HID);
    k_cvt_bf16<<<eblocks((size_t)FEAT * CATK), 256, 0, stream>>>(stW0, stW0b, FEAT * CATK);
    k_cvt_bf16<<<eblocks((size_t)FEAT * FEAT), 256, 0, stream>>>(stW1, stW1b, FEAT * FEAT);
    k_cvt_bf16<<<eblocks((size_t)2 * STOCH * FEAT), 256, 0, stream>>>(stW2, stW2b, 2 * STOCH * FEAT);
    k_cvt_bf16<<<eblocks((size_t)FEAT * STATE), 256, 0, stream>>>(epW0, epW0b, FEAT * STATE);
    k_cvt_bf16<<<eblocks((size_t)FEAT * FEAT), 256, 0, stream>>>(epW1, epW1b, FEAT * FEAT);
    k_cvt_bf16<<<eblocks((size_t)EMB_ * FEAT), 256, 0, stream>>>(epW2, epW2b, EMB_ * FEAT);
    k_cvt_bf16<<<eblocks((size_t)FEAT * STATE), 256, 0, stream>>>(rpW0, rpW0b, FEAT * STATE);
    k_cvt_bf16<<<eblocks((size_t)FEAT * FEAT), 256, 0, stream>>>(rpW1, rpW1b, FEAT * FEAT);

    k_init_state<<<eblocks((size_t)BB * HID), 256, 0, stream>>>(h, hbf, s, sbf, acc);

    // --- sequential scan over T ---
    for (int t = 0; t < TT; ++t) {
        const float* at   = action + (size_t)t * BB * ACT_;
        const float* ep_  = emb + (size_t)(t > 0 ? t - 1 : 0) * BB * EMB_;
        const float* epst = eps + (size_t)t * BB * STOCH;
        float* stF = statesF + (size_t)t * BB * STATE;
        bf16*  stB = statesB + (size_t)t * BB * STATE;

        k_build_xin<<<eblocks((size_t)BB * GK), 256, 0, stream>>>(s, at, xinb);
        k_build_hscat<<<eblocks((size_t)BB * STATE), 256, 0, stream>>>(h, s, hscat);
        k_copy_emb<<<eblocks((size_t)BB * EMB_), 256, 0, stream>>>(ep_, catQ + HID);

        // GRU
        gemm4(xinb, wihp, gbih, gi, 3 * HID, nullptr, 0, BB, 3 * HID, GK, 0);
        gemm4(hbf, whhb, gbhh, gh, 3 * HID, nullptr, 0, BB, 3 * HID, HID, 0);
        k_gru_gate<<<eblocks((size_t)BB * HID), 256, 0, stream>>>(
            gi, gh, h, hbf, catP, catQ, stF, stB);

        // pre_emb = ep-MLP([h_prev | s_prev]) -> catP[:, 512:1536]
        gemm4(hscat, epW0b, epb0, nullptr, 0, t0b, FEAT, BB, FEAT, STATE, 1);
        gemm4(t0b, epW1b, epb1, nullptr, 0, t1b, FEAT, BB, FEAT, FEAT, 1);
        gemm4(t1b, epW2b, epb2, nullptr, 0, catP + HID, CATK, BB, EMB_, FEAT, 0);

        // prior = st-MLP([next_h | pre_emb])
        gemm4(catP, stW0b, stb0, nullptr, 0, u0b, FEAT, BB, FEAT, CATK, 1);
        gemm4(u0b, stW1b, stb1, nullptr, 0, u1b, FEAT, BB, FEAT, FEAT, 1);
        gemm1(u1b, stW2b, stb2, praw, 2 * STOCH, nullptr, 0, BB, 2 * STOCH, FEAT, 0);

        // posterior = st-MLP([next_h | emb_prev])
        gemm4(catQ, stW0b, stb0, nullptr, 0, v0b, FEAT, BB, FEAT, CATK, 1);
        gemm4(v0b, stW1b, stb1, nullptr, 0, v1b, FEAT, BB, FEAT, FEAT, 1);
        gemm1(v1b, stW2b, stb2, qraw, 2 * STOCH, nullptr, 0, BB, 2 * STOCH, FEAT, 0);

        k_sample_kl<<<eblocks((size_t)BB * STOCH), 256, 0, stream>>>(
            qraw, praw, epst, s, sbf, stF, stB, acc + 0);
    }

    // --- final heads over all T*B rows, in chunks ---
    for (int ch = 0; ch < NROW / CH; ++ch) {
        const bf16* sb = statesB + (size_t)ch * CH * STATE;
        gemm4(sb, epW0b, epb0, nullptr, 0, e0b, FEAT, CH, FEAT, STATE, 1);
        gemm4(e0b, epW1b, epb1, nullptr, 0, e1b, FEAT, CH, FEAT, FEAT, 1);
        gemm4(e1b, epW2b, epb2, embmu, EMB_, nullptr, 0, CH, EMB_, FEAT, 0);
        k_sqloss<<<eblocks((size_t)CH * EMB_), 256, 0, stream>>>(
            embmu, emb + (size_t)ch * CH * EMB_, CH * EMB_, acc + 1);
        gemm4(sb, rpW0b, rpb0, nullptr, 0, r0b, FEAT, CH, FEAT, STATE, 1);
        gemm4(r0b, rpW1b, rpb1, r1f, FEAT, nullptr, 0, CH, FEAT, FEAT, 1);
        k_rhead<<<eblocks((size_t)CH), 256, 0, stream>>>(
            r1f, rpW2, rpb2, reward + (size_t)ch * CH, CH, acc + 2);
    }

    k_final<<<1, 1, 0, stream>>>(acc, (float*)d_out);
}
